// OctreeAttention_20109036880176
// MI455X (gfx1250) — compile-verified, matrix operand order checked by reading, not run on hardware
//
#include <hip/hip_runtime.h>
#include <cstdint>
#include <cstddef>

// ---------------------------------------------------------------------------
// OctreeAttention for MI455X (gfx1250, wave32, WMMA).
// Pipeline: cast/pad -> bf16 WMMA GEMM (qkv) -> windowed WMMA attention ->
//           bf16 WMMA GEMM (proj, fp32 out).
// ---------------------------------------------------------------------------

typedef unsigned short ushort_t;
typedef __attribute__((ext_vector_type(16))) __bf16 v16bf;
typedef __attribute__((ext_vector_type(8)))  float  v8f;
typedef __attribute__((ext_vector_type(4)))  unsigned int v4u;

constexpr int   kN   = 65529;    // tokens
constexpr int   kC   = 1024;     // channels
constexpr int   kNP  = 65536;    // padded tokens (multiple of 128)
constexpr int   kQKV = 3072;     // 3*C
constexpr float kScale = 0.125f; // (C/H)^-0.5 = 64^-0.5
constexpr float kNeg   = -1000.0f;

__device__ __forceinline__ ushort_t f2bf(float f) {
  unsigned u = __float_as_uint(f);
  u += 0x7FFFu + ((u >> 16) & 1u);      // round-to-nearest-even
  return (ushort_t)(u >> 16);
}

union Frag16 { v16bf v; v4u q[2]; };    // 8 VGPRs: two 16B chunks per lane
union FAcc   { v8f  v; float f[8]; };   // 16x16 f32 C/D tile per lane

// ---------------------------------------------------------------------------
// Cast fp32 data -> bf16, pad rows [kN, kNP) with zeros.
// ---------------------------------------------------------------------------
__global__ __launch_bounds__(256) void cast_pad_data(
    const float* __restrict__ src, ushort_t* __restrict__ dst) {
  long e0 = ((long)blockIdx.x * blockDim.x + threadIdx.x) * 8;
  if (e0 >= (long)kNP * kC) return;
  long row = e0 / kC;
  if (row < kN) {
#pragma unroll
    for (int i = 0; i < 8; ++i) dst[e0 + i] = f2bf(src[e0 + i]);
  } else {
#pragma unroll
    for (int i = 0; i < 8; ++i) dst[e0 + i] = 0;
  }
}

// ---------------------------------------------------------------------------
// Transpose + cast weights: src[K x Nc] fp32 -> dst[Nc x K] bf16.
// Contiguous writes; strided reads (weights are small, L2-resident).
// ---------------------------------------------------------------------------
__global__ __launch_bounds__(256) void transpose_cast(
    const float* __restrict__ src, ushort_t* __restrict__ dst, int K, int Nc) {
  long idx = (long)blockIdx.x * blockDim.x + threadIdx.x;
  if (idx >= (long)K * Nc) return;
  long n = idx / K, k = idx % K;
  dst[idx] = f2bf(src[k * (long)Nc + n]);
}

// ---------------------------------------------------------------------------
// bf16 WMMA GEMM: C[M x Nn] = A[M x Kd] * Bt[Nn x Kd]^T + bias.
// Block tile 128x128x32, 8 waves (2x4), each wave 64x32 (8 WMMA tiles).
// LDS stride 40 ushorts (80B) -> conflict-free b128 fragment loads.
// ---------------------------------------------------------------------------
template <bool OUT_BF16>
__global__ __launch_bounds__(256) void gemm_bf16(
    const ushort_t* __restrict__ A, const ushort_t* __restrict__ Bt,
    const float* __restrict__ bias, void* __restrict__ Cout,
    int Kd, int Nn, int rowLimit) {
  constexpr int LDT = 40;
  __shared__ __align__(16) ushort_t sA[128 * LDT];
  __shared__ __align__(16) ushort_t sB[128 * LDT];

  const int tid  = threadIdx.x;
  const int lane = tid & 31, wave = tid >> 5;
  const int wrow = wave >> 2, wcol = wave & 3;   // 2x4 wave grid
  const int lm = lane & 15, lh = lane >> 4;
  const long m0 = (long)blockIdx.y * 128;
  const long n0 = (long)blockIdx.x * 128;

  v8f acc[4][2] = {};

  for (int k0 = 0; k0 < Kd; k0 += 32) {
    // Stage 128x32 bf16 tiles of A and Bt (contiguous b128 loads).
#pragma unroll
    for (int c = tid; c < 512; c += 256) {
      int r = c >> 2, s = (c & 3) * 8;
      *(v4u*)&sA[r * LDT + s] = *(const v4u*)&A[(m0 + r) * Kd + k0 + s];
      *(v4u*)&sB[r * LDT + s] = *(const v4u*)&Bt[(n0 + r) * Kd + k0 + s];
    }
    __syncthreads();

    Frag16 a[4], b[2];
#pragma unroll
    for (int mt = 0; mt < 4; ++mt) {
      int r = wrow * 64 + mt * 16 + lm;
      a[mt].q[0] = *(const v4u*)&sA[r * LDT + 8 * lh];
      a[mt].q[1] = *(const v4u*)&sA[r * LDT + 16 + 8 * lh];
    }
#pragma unroll
    for (int nt = 0; nt < 2; ++nt) {
      int r = wcol * 32 + nt * 16 + lm;
      b[nt].q[0] = *(const v4u*)&sB[r * LDT + 8 * lh];
      b[nt].q[1] = *(const v4u*)&sB[r * LDT + 16 + 8 * lh];
    }
#pragma unroll
    for (int mt = 0; mt < 4; ++mt)
#pragma unroll
      for (int nt = 0; nt < 2; ++nt)
        acc[mt][nt] = __builtin_amdgcn_wmma_f32_16x16x32_bf16(
            false, a[mt].v, false, b[nt].v, (short)0, acc[mt][nt], false, false);
    __syncthreads();
  }

  // Epilogue: bias add, write bf16 or fp32 (row-guarded for the final GEMM).
#pragma unroll
  for (int mt = 0; mt < 4; ++mt)
#pragma unroll
    for (int nt = 0; nt < 2; ++nt) {
      long j = n0 + wcol * 32 + nt * 16 + lm;
      float bj = bias[j];
      FAcc t; t.v = acc[mt][nt];
#pragma unroll
      for (int vg = 0; vg < 8; ++vg) {
        long i = m0 + wrow * 64 + mt * 16 + vg + 8 * lh;
        if (i < rowLimit) {
          float val = t.f[vg] + bj;
          if (OUT_BF16) ((ushort_t*)Cout)[i * Nn + j] = f2bf(val);
          else          ((float*)Cout)[i * Nn + j]    = val;
        }
      }
    }
}

// ---------------------------------------------------------------------------
// Windowed attention. One workgroup per window (2048 windows of 32 tokens),
// one wave per head (2 heads per wave, 16 heads). Dilation: token kk of
// window w lives at global row (w/4)*128 + kk*4 + (w%4).
// ---------------------------------------------------------------------------
__global__ __launch_bounds__(256) void attn_kernel(
    const ushort_t* __restrict__ qkv,   // kNP x 3072 bf16
    const int* __restrict__ batch, const int* __restrict__ depth,
    ushort_t* __restrict__ outp) {      // kNP x 1024 bf16
  constexpr int LDP = 40;
  __shared__ int sb[32], sd[32];
  __shared__ __align__(16) ushort_t sP[8][32 * LDP];   // softmax(P) per wave
  __shared__ __align__(16) ushort_t sVT[8][64 * LDP];  // V^T per wave

  const int w = blockIdx.x;
  const long rbase = (long)(w >> 2) * 128 + (w & 3);
  const int tid = threadIdx.x, lane = tid & 31, wave = tid >> 5;
  const int lm = lane & 15, lh = lane >> 4;

  if (tid < 32) {
    long ro = rbase + tid * 4;
    sb[tid] = (ro < kN) ? batch[ro] : -1;
    sd[tid] = (ro < kN) ? depth[ro] : 0x7fffffff;
  }
  __syncthreads();

  ushort_t* P  = sP[wave];
  ushort_t* VT = sVT[wave];

  for (int hi = 0; hi < 2; ++hi) {
    const int h = wave + hi * 8;

    // Stage V^T into per-wave LDS (lane = token j, scatter b16 per channel).
    {
      const ushort_t* vp = qkv + (rbase + lane * 4) * kQKV + 2048 + h * 64;
#pragma unroll
      for (int cc = 0; cc < 64; cc += 8) {
        union { v4u q; ushort_t e[8]; } x;
        x.q = *(const v4u*)(vp + cc);
#pragma unroll
        for (int e = 0; e < 8; ++e) VT[(cc + e) * LDP + lane] = x.e[e];
      }
    }

    // Q (A-frag) and K (B-frag) straight from global: token-major rows match
    // the WMMA 16-bit lane layout; two b128 chunks per fragment per lane.
    Frag16 qa[2][2], kb[2][2];
#pragma unroll
    for (int t = 0; t < 2; ++t) {
      long ro = rbase + (t * 16 + lm) * 4;
      const ushort_t* qp = qkv + ro * kQKV + h * 64;
      const ushort_t* kp = qkv + ro * kQKV + 1024 + h * 64;
#pragma unroll
      for (int kt = 0; kt < 2; ++kt) {
        qa[t][kt].q[0] = *(const v4u*)(qp + kt * 32 + 8 * lh);
        qa[t][kt].q[1] = *(const v4u*)(qp + kt * 32 + 16 + 8 * lh);
        kb[t][kt].q[0] = *(const v4u*)(kp + kt * 32 + 8 * lh);
        kb[t][kt].q[1] = *(const v4u*)(kp + kt * 32 + 16 + 8 * lh);
      }
    }

    // attn = Q K^T : 2x2 tiles, 2 K-steps of 32 channels each.
    v8f c[2][2] = {};
#pragma unroll
    for (int mt = 0; mt < 2; ++mt)
#pragma unroll
      for (int nt = 0; nt < 2; ++nt)
#pragma unroll
        for (int kt = 0; kt < 2; ++kt)
          c[mt][nt] = __builtin_amdgcn_wmma_f32_16x16x32_bf16(
              false, qa[mt][kt].v, false, kb[nt][kt].v, (short)0, c[mt][nt],
              false, false);

    // Scale + mask + row softmax in C-tile layout. Row i is spread over the
    // 16 lanes of one half (shfl_xor masks 1..8 stay within a half).
#pragma unroll
    for (int mt = 0; mt < 2; ++mt) {
      FAcc t0, t1; t0.v = c[mt][0]; t1.v = c[mt][1];
#pragma unroll
      for (int vg = 0; vg < 8; ++vg) {
        int i  = mt * 16 + vg + 8 * lh;
        int j0 = lm, j1 = 16 + lm;
        float m0v = (sb[i] != sb[j0] ? kNeg : 0.0f) + (sd[i] < sd[j0] ? kNeg : 0.0f);
        float m1v = (sb[i] != sb[j1] ? kNeg : 0.0f) + (sd[i] < sd[j1] ? kNeg : 0.0f);
        float a0 = t0.f[vg] * kScale + m0v;
        float a1 = t1.f[vg] * kScale + m1v;
        float mx = fmaxf(a0, a1);
#pragma unroll
        for (int s = 8; s >= 1; s >>= 1) mx = fmaxf(mx, __shfl_xor(mx, s, 32));
        float x0 = __expf(a0 - mx), x1 = __expf(a1 - mx);
        float sm = x0 + x1;
#pragma unroll
        for (int s = 8; s >= 1; s >>= 1) sm += __shfl_xor(sm, s, 32);
        float inv = 1.0f / sm;
        P[i * LDP + j0] = f2bf(x0 * inv);
        P[i * LDP + j1] = f2bf(x1 * inv);
      }
    }

    // out = P V : A-frags from P (LDS), B-frags from V^T (LDS); K-dim = 32.
    Frag16 pa[2], vb[4];
#pragma unroll
    for (int mt = 0; mt < 2; ++mt) {
      int r = mt * 16 + lm;
      pa[mt].q[0] = *(const v4u*)&P[r * LDP + 8 * lh];
      pa[mt].q[1] = *(const v4u*)&P[r * LDP + 16 + 8 * lh];
    }
#pragma unroll
    for (int nt = 0; nt < 4; ++nt) {
      int ch = nt * 16 + lm;
      vb[nt].q[0] = *(const v4u*)&VT[ch * LDP + 8 * lh];
      vb[nt].q[1] = *(const v4u*)&VT[ch * LDP + 16 + 8 * lh];
    }
#pragma unroll
    for (int mt = 0; mt < 2; ++mt)
#pragma unroll
      for (int nt = 0; nt < 4; ++nt) {
        v8f o = {};
        o = __builtin_amdgcn_wmma_f32_16x16x32_bf16(
            false, pa[mt].v, false, vb[nt].v, (short)0, o, false, false);
        FAcc t; t.v = o;
        int ch = h * 64 + nt * 16 + lm;
#pragma unroll
        for (int vg = 0; vg < 8; ++vg) {
          int i = mt * 16 + vg + 8 * lh;
          outp[(rbase + i * 4) * kC + ch] = f2bf(t.f[vg]);  // undo dilation
        }
      }
  }
}

// ---------------------------------------------------------------------------
// Launch
// ---------------------------------------------------------------------------
extern "C" void kernel_launch(void* const* d_in, const int* in_sizes, int n_in,
                              void* d_out, int out_size, void* d_ws,
                              size_t ws_size, hipStream_t stream) {
  const float* data   = (const float*)d_in[0];
  const float* w_qkv  = (const float*)d_in[1];
  const float* b_qkv  = (const float*)d_in[2];
  const float* w_proj = (const float*)d_in[3];
  const float* b_proj = (const float*)d_in[4];
  const int*   batch  = (const int*)d_in[5];
  const int*   depth  = (const int*)d_in[6];
  float* out = (float*)d_out;

  char* ws = (char*)d_ws;
  ushort_t* data_bf = (ushort_t*)ws;  ws += (size_t)kNP * kC * 2;
  ushort_t* wqkvT   = (ushort_t*)ws;  ws += (size_t)kQKV * kC * 2;
  ushort_t* wprojT  = (ushort_t*)ws;  ws += (size_t)kC * kC * 2;
  ushort_t* qkv     = (ushort_t*)ws;  ws += (size_t)kNP * kQKV * 2;
  ushort_t* aout    = (ushort_t*)ws;

  cast_pad_data<<<(int)(((long)kNP * kC / 8 + 255) / 256), 256, 0, stream>>>(
      data, data_bf);
  transpose_cast<<<(kC * kQKV + 255) / 256, 256, 0, stream>>>(
      w_qkv, wqkvT, kC, kQKV);
  transpose_cast<<<(kC * kC + 255) / 256, 256, 0, stream>>>(
      w_proj, wprojT, kC, kC);

  gemm_bf16<true><<<dim3(kQKV / 128, kNP / 128), 256, 0, stream>>>(
      data_bf, wqkvT, b_qkv, qkv, kC, kQKV, kNP);

  attn_kernel<<<kNP / 32, 256, 0, stream>>>(qkv, batch, depth, aout);

  gemm_bf16<false><<<dim3(kC / 128, kNP / 128), 256, 0, stream>>>(
      aout, wprojT, b_proj, out, kC, kC, kN);
}
